// MessagePassingLayerEW_17471926960851
// MI455X (gfx1250) — matrix operand order repaired and checked by reading, not compiled, not run on hardware
//
#include <hip/hip_runtime.h>
#include <hip/hip_bf16.h>

typedef __attribute__((ext_vector_type(16))) _Float16 v16h;
typedef __attribute__((ext_vector_type(8)))  _Float16 v8h;
typedef __attribute__((ext_vector_type(4)))  _Float16 v4h;
typedef __attribute__((ext_vector_type(8)))  float    v8f;

#define N_GRAPHS  50
#define NODES_PER 1000
#define EDGES_PER 8000
#define N_NODES   50000
#define N_EDGES   400000
#define ND 64
#define ED 32
#define GD 32
#define CONCAT 192            // 2*ND + ED + GD

#define TILE_E 64             // edges per block in the fused MLP kernel

// d_out layout (floats): new_nodes | new_edges | new_global
#define OUT_EDGES_OFF (N_NODES * ND)                   // 3,200,000
#define OUT_GLOB_OFF  (OUT_EDGES_OFF + N_EDGES * ED)   // 16,000,000

// ---------------------------------------------------------------------------
// WMMA fragment helpers (CDNA5 v_wmma_f32_16x16x32_f16; layouts per ISA 7.12.2)
//   A (16x32 f16): lane m=lane&15, hi=lane>>4; halves e=0..7 -> K=k0+hi*8+e,
//                  e=8..15 -> K=k0+16+hi*8+(e-8)  (two contiguous 16B chunks)
//   B (32x16 f16): lane col=lane&15; half e -> K=k0+hi*8+(e&7)+(e>>3)*16
//   C (16x16 f32): VGPR v -> M=v+8*hi, N=lane&15
// B fragments are PRE-SWIZZLED in ws: packed[(nt*Kt+kt)*32+lane] = 16 halves,
// so each lane reads one contiguous 32-byte chunk (2 x global_load_b128).
// ---------------------------------------------------------------------------
__device__ __forceinline__ v16h load_a_frag(const _Float16* __restrict__ A, int lda,
                                            int row, int k0, int hi) {
    const v8h lo = *(const v8h*)&A[row * lda + k0 + hi * 8];
    const v8h hh = *(const v8h*)&A[row * lda + k0 + 16 + hi * 8];
    return __builtin_shufflevector(lo, hh, 0, 1, 2, 3, 4, 5, 6, 7,
                                           8, 9, 10, 11, 12, 13, 14, 15);
}

__device__ __forceinline__ v8f wmma_f16(v16h a, v16h b, v8f c) {
    return __builtin_amdgcn_wmma_f32_16x16x32_f16(
        /*neg_a=*/false, a, /*neg_b=*/false, b,
        /*c_mod=*/(short)0, c, /*reuse_a=*/false, /*reuse_b=*/false);
}

template<int LDH>
__device__ __forceinline__ void store_relu_f16(_Float16* __restrict__ H, int row0, int col0,
                                               v8f acc, const float* __restrict__ bias) {
    const int lane = threadIdx.x & 31;
    const int mr   = lane & 15;
    const int hi   = lane >> 4;
    const float bv = bias[col0 + mr];
#pragma unroll
    for (int v = 0; v < 8; ++v) {
        float x = acc[v] + bv;
        x = x > 0.f ? x : 0.f;
        H[(row0 + v + hi * 8) * LDH + col0 + mr] = (_Float16)x;
    }
}

// ---------------------------------------------------------------------------
// Kernel 0: zero new_nodes (scatter target) and the ws segment-sum buffers
// ---------------------------------------------------------------------------
__global__ void zero_kernel(float* __restrict__ outNodes, float* __restrict__ wsSums) {
    const int stride = gridDim.x * blockDim.x;
    for (int i = blockIdx.x * blockDim.x + threadIdx.x; i < N_NODES * ND; i += stride)
        outNodes[i] = 0.f;
    for (int i = blockIdx.x * blockDim.x + threadIdx.x; i < N_GRAPHS * (ND + ED); i += stride)
        wsSums[i] = 0.f;
}

// ---------------------------------------------------------------------------
// Kernel 1: convert + swizzle weights f32 -> f16 WMMA-B-fragment order.
// W is K x N row-major. Packed: P[((nt*Kt+kt)*32+lane)*16+e] =
//   W[(kt*32 + (lane>>4)*8 + (e&7) + (e>>3)*16) * N + nt*16 + (lane&15)]
// ---------------------------------------------------------------------------
__device__ __forceinline__ void pack_wmma_b(const float* __restrict__ W, _Float16* __restrict__ P,
                                            int Kt, int Nt, int N, int tid, int stride) {
    const int total = Kt * Nt * 512;
    for (int t = tid; t < total; t += stride) {
        const int e    = t & 15;
        const int lane = (t >> 4) & 31;
        const int rem  = t >> 9;           // nt*Kt + kt
        const int kt   = rem % Kt;
        const int nt   = rem / Kt;
        const int k = kt * 32 + (lane >> 4) * 8 + (e & 7) + (e >> 3) * 16;
        const int n = nt * 16 + (lane & 15);
        P[t] = (_Float16)W[k * N + n];
    }
}

__global__ void prep_weights(const float* __restrict__ w1n, const float* __restrict__ w2n,
                             const float* __restrict__ w1e, const float* __restrict__ w2e,
                             _Float16* __restrict__ p1n, _Float16* __restrict__ p2n,
                             _Float16* __restrict__ p1e, _Float16* __restrict__ p2e) {
    const int stride = gridDim.x * blockDim.x;
    const int tid = blockIdx.x * blockDim.x + threadIdx.x;
    pack_wmma_b(w1n, p1n, 6, 8, 128, tid, stride);   // 192x128
    pack_wmma_b(w2n, p2n, 4, 4, 64,  tid, stride);   // 128x64
    pack_wmma_b(w1e, p1e, 6, 4, 64,  tid, stride);   // 192x64
    pack_wmma_b(w2e, p2e, 2, 2, 32,  tid, stride);   // 64x32
}

// ---------------------------------------------------------------------------
// Kernel 2: fused per-edge node-MLP + edge-MLP with WMMA.
// Block = 256 threads (8 wave32s), TILE_E = 64 edges.
// Each wave owns an N-strip; B fragments cached in registers across m-tiles.
//   stage B: node hidden nt=wave (4 mt);   edge hidden nt=wave>>1, 2 mt
//   stage C: node out nt=wave&3, 2 mt;     edge out nt=wave&1, mt=wave>>1
// ---------------------------------------------------------------------------
__global__ __launch_bounds__(256) void edge_mlp_kernel(
    const float* __restrict__ nodes, const float* __restrict__ edges,
    const float* __restrict__ globals_, const float* __restrict__ ew,
    const int* __restrict__ senders, const int* __restrict__ receivers,
    const _Float16* __restrict__ pw1n, const float* __restrict__ b1n,
    const _Float16* __restrict__ pw2n, const float* __restrict__ b2n,
    const _Float16* __restrict__ pw1e, const float* __restrict__ b1e,
    const _Float16* __restrict__ pw2e, const float* __restrict__ b2e,
    float* __restrict__ outNodes, float* __restrict__ outEdges) {

    __shared__ _Float16 sX [TILE_E * CONCAT];   // 24 KB
    __shared__ _Float16 sHn[TILE_E * 128];      // 16 KB
    __shared__ _Float16 sHe[TILE_E * 64];       //  8 KB
    __shared__ float sB1n[128], sB1e[64], sB2n[64], sB2e[32];

    const int tid = threadIdx.x;
    const int e0  = blockIdx.x * TILE_E;

    // stage A: build weighted concat tile in LDS (f16), float4-vectorized
    for (int i = tid; i < TILE_E * (CONCAT / 4); i += 256) {
        const int r  = i / (CONCAT / 4);
        const int c4 = (i - r * (CONCAT / 4)) * 4;
        const int e  = e0 + r;
        const float w = ew[e];
        float4 v;
        if (c4 < ND)               v = *(const float4*)&nodes[senders[e] * ND + c4];
        else if (c4 < 2 * ND)      v = *(const float4*)&nodes[receivers[e] * ND + c4 - ND];
        else if (c4 < 2 * ND + ED) v = *(const float4*)&edges[e * ED + c4 - 2 * ND];
        else                       v = *(const float4*)&globals_[(e / EDGES_PER) * GD + c4 - 2 * ND - ED];
        v4h hv = { (_Float16)(w * v.x), (_Float16)(w * v.y),
                   (_Float16)(w * v.z), (_Float16)(w * v.w) };
        *(v4h*)&sX[r * CONCAT + c4] = hv;
    }
    if (tid < 128) sB1n[tid] = b1n[tid];
    if (tid < 64)  { sB1e[tid] = b1e[tid]; sB2n[tid] = b2n[tid]; }
    if (tid < 32)  sB2e[tid] = b2e[tid];
    __syncthreads();

    const int wave = tid >> 5;
    const int lane = tid & 31;
    const int mr   = lane & 15;
    const int hi   = lane >> 4;

    const v16h* Bp1n = (const v16h*)pw1n;
    const v16h* Bp2n = (const v16h*)pw2n;
    const v16h* Bp1e = (const v16h*)pw1e;
    const v16h* Bp2e = (const v16h*)pw2e;

    // ---- stage B: node hidden (K=192, N-strip nt=wave, 4 m-tiles) ----
    {
        const int nt = wave;
        v16h B[6];
#pragma unroll
        for (int kt = 0; kt < 6; ++kt) B[kt] = Bp1n[(nt * 6 + kt) * 32 + lane];
#pragma unroll
        for (int mt = 0; mt < 4; ++mt) {
            v8f acc = {};
#pragma unroll
            for (int kt = 0; kt < 6; ++kt) {
                v16h a = load_a_frag(sX, CONCAT, mt * 16 + mr, kt * 32, hi);
                acc = wmma_f16(a, B[kt], acc);
            }
            store_relu_f16<128>(sHn, mt * 16, nt * 16, acc, sB1n);
        }
    }
    // ---- stage B: edge hidden (K=192, nt=wave>>1, 2 m-tiles) ----
    {
        const int nt = wave >> 1;
        v16h B[6];
#pragma unroll
        for (int kt = 0; kt < 6; ++kt) B[kt] = Bp1e[(nt * 6 + kt) * 32 + lane];
#pragma unroll
        for (int m = 0; m < 2; ++m) {
            const int mt = (wave & 1) * 2 + m;
            v8f acc = {};
#pragma unroll
            for (int kt = 0; kt < 6; ++kt) {
                v16h a = load_a_frag(sX, CONCAT, mt * 16 + mr, kt * 32, hi);
                acc = wmma_f16(a, B[kt], acc);
            }
            store_relu_f16<64>(sHe, mt * 16, nt * 16, acc, sB1e);
        }
    }
    __syncthreads();

    // ---- stage C: node out (K=128, nt=wave&3, 2 m-tiles) -> scatter add ----
    {
        const int nt = wave & 3;
        v16h B[4];
#pragma unroll
        for (int kt = 0; kt < 4; ++kt) B[kt] = Bp2n[(nt * 4 + kt) * 32 + lane];
        const int col = nt * 16 + mr;
        const float bv = sB2n[col];
#pragma unroll
        for (int m = 0; m < 2; ++m) {
            const int mt = (wave >> 2) * 2 + m;
            v8f acc = {};
#pragma unroll
            for (int kt = 0; kt < 4; ++kt) {
                v16h a = load_a_frag(sHn, 128, mt * 16 + mr, kt * 32, hi);
                acc = wmma_f16(a, B[kt], acc);
            }
#pragma unroll
            for (int v = 0; v < 8; ++v) {
                const int ge = e0 + mt * 16 + v + hi * 8;
                atomicAdd(&outNodes[receivers[ge] * ND + col], acc[v] + bv);
            }
        }
    }
    // ---- stage C: edge out (K=64, nt=wave&1, mt=wave>>1) -> direct store ----
    {
        const int nt = wave & 1;
        const int mt = wave >> 1;
        v16h B[2];
#pragma unroll
        for (int kt = 0; kt < 2; ++kt) B[kt] = Bp2e[(nt * 2 + kt) * 32 + lane];
        v8f acc = {};
#pragma unroll
        for (int kt = 0; kt < 2; ++kt) {
            v16h a = load_a_frag(sHe, 64, mt * 16 + mr, kt * 32, hi);
            acc = wmma_f16(a, B[kt], acc);
        }
        const int col = nt * 16 + mr;
        const float bv = sB2e[col];
#pragma unroll
        for (int v = 0; v < 8; ++v) {
            const int ge = e0 + mt * 16 + v + hi * 8;
            outEdges[ge * ED + col] = acc[v] + bv;
        }
    }
}

// ---------------------------------------------------------------------------
// Kernel 3: per-graph feature sums (contiguous segments, block reduce)
// ---------------------------------------------------------------------------
__global__ void sum_nodes_kernel(const float* __restrict__ nodes, float* __restrict__ sumN) {
    const int g = blockIdx.x, t = threadIdx.x;
    const int c = t & 63, rg = t >> 6;           // 4 row groups
    float acc = 0.f;
    for (int r = rg; r < NODES_PER; r += 4)
        acc += nodes[(g * NODES_PER + r) * ND + c];
    __shared__ float s[256];
    s[t] = acc;
    __syncthreads();
    if (t < 64) sumN[g * ND + c] = s[c] + s[c + 64] + s[c + 128] + s[c + 192];
}

__global__ void sum_edges_kernel(const float* __restrict__ edges, const float* __restrict__ ew,
                                 float* __restrict__ sumE) {
    const int g = blockIdx.x, t = threadIdx.x;
    const int c = t & 31, rg = t >> 5;           // 8 row groups
    float acc = 0.f;
    for (int r = rg; r < EDGES_PER; r += 8) {
        const int e = g * EDGES_PER + r;
        acc += ew[e] * edges[e * ED + c];
    }
    __shared__ float s[256];
    s[t] = acc;
    __syncthreads();
    if (t < 32) {
        float v = 0.f;
#pragma unroll
        for (int k = 0; k < 8; ++k) v += s[c + k * 32];
        sumE[g * ED + c] = v;
    }
}

// ---------------------------------------------------------------------------
// Kernel 4: per-graph global update (50 rows; tiny, scalar f32)
// ---------------------------------------------------------------------------
__global__ void globals_kernel(
    const float* __restrict__ sumN, const float* __restrict__ sumE,
    const float* __restrict__ globals_,
    const float* gnW1, const float* gnb1, const float* gnW2, const float* gnb2,
    const float* geW1, const float* geb1, const float* geW2, const float* geb2,
    const float* ggW1, const float* ggb1, const float* ggW2, const float* ggb2,
    const float* gfW1, const float* gfb1, const float* gfW2, const float* gfb2,
    float* __restrict__ outG) {
    const int g = blockIdx.x, t = threadIdx.x;  // 64 threads
    __shared__ float xin[96], h[64], cat[96];

    // gnode: 64 -> 64 relu -> 32  (tn at cat[32..63])
    if (t < 64) xin[t] = sumN[g * ND + t];
    __syncthreads();
    { float a = gnb1[t]; for (int k = 0; k < 64; ++k) a += xin[k] * gnW1[k * 64 + t];
      h[t] = a > 0.f ? a : 0.f; }
    __syncthreads();
    if (t < 32) { float a = gnb2[t]; for (int k = 0; k < 64; ++k) a += h[k] * gnW2[k * 32 + t];
                  cat[32 + t] = a; }
    __syncthreads();

    // gedge: 32 -> 64 relu -> 32  (te at cat[64..95])
    if (t < 32) xin[t] = sumE[g * ED + t];
    __syncthreads();
    { float a = geb1[t]; for (int k = 0; k < 32; ++k) a += xin[k] * geW1[k * 64 + t];
      h[t] = a > 0.f ? a : 0.f; }
    __syncthreads();
    if (t < 32) { float a = geb2[t]; for (int k = 0; k < 64; ++k) a += h[k] * geW2[k * 32 + t];
                  cat[64 + t] = a; }
    __syncthreads();

    // gglob: 32 -> 64 relu -> 32  (tg at cat[0..31])
    if (t < 32) xin[t] = globals_[g * GD + t];
    __syncthreads();
    { float a = ggb1[t]; for (int k = 0; k < 32; ++k) a += xin[k] * ggW1[k * 64 + t];
      h[t] = a > 0.f ? a : 0.f; }
    __syncthreads();
    if (t < 32) { float a = ggb2[t]; for (int k = 0; k < 64; ++k) a += h[k] * ggW2[k * 32 + t];
                  cat[t] = a; }
    __syncthreads();

    // gfinal: 96 -> 64 relu -> 32
    { float a = gfb1[t]; for (int k = 0; k < 96; ++k) a += cat[k] * gfW1[k * 64 + t];
      h[t] = a > 0.f ? a : 0.f; }
    __syncthreads();
    if (t < 32) { float a = gfb2[t]; for (int k = 0; k < 64; ++k) a += h[k] * gfW2[k * 32 + t];
                  outG[g * GD + t] = a; }
}

// ---------------------------------------------------------------------------
// Launch. Params assumed flattened in JAX pytree (sorted-key) order:
//   edge(W1,b1,W2,b2), gedge(...), gfinal(...), gglob(...), gnode(...), node(...)
// ---------------------------------------------------------------------------
extern "C" void kernel_launch(void* const* d_in, const int* in_sizes, int n_in,
                              void* d_out, int out_size, void* d_ws, size_t ws_size,
                              hipStream_t stream) {
    const float* nodes    = (const float*)d_in[0];
    const float* edges    = (const float*)d_in[1];
    const float* globals_ = (const float*)d_in[2];
    const float* ew       = (const float*)d_in[3];

    const float* eW1 = (const float*)d_in[4];   const float* eb1 = (const float*)d_in[5];
    const float* eW2 = (const float*)d_in[6];   const float* eb2 = (const float*)d_in[7];
    const float* geW1 = (const float*)d_in[8];  const float* geb1 = (const float*)d_in[9];
    const float* geW2 = (const float*)d_in[10]; const float* geb2 = (const float*)d_in[11];
    const float* gfW1 = (const float*)d_in[12]; const float* gfb1 = (const float*)d_in[13];
    const float* gfW2 = (const float*)d_in[14]; const float* gfb2 = (const float*)d_in[15];
    const float* ggW1 = (const float*)d_in[16]; const float* ggb1 = (const float*)d_in[17];
    const float* ggW2 = (const float*)d_in[18]; const float* ggb2 = (const float*)d_in[19];
    const float* gnW1 = (const float*)d_in[20]; const float* gnb1 = (const float*)d_in[21];
    const float* gnW2 = (const float*)d_in[22]; const float* gnb2 = (const float*)d_in[23];
    const float* nW1 = (const float*)d_in[24];  const float* nb1 = (const float*)d_in[25];
    const float* nW2 = (const float*)d_in[26];  const float* nb2 = (const float*)d_in[27];

    const int* senders   = (const int*)d_in[28];
    const int* receivers = (const int*)d_in[29];

    float* out      = (float*)d_out;
    float* outNodes = out;
    float* outEdges = out + OUT_EDGES_OFF;
    float* outG     = out + OUT_GLOB_OFF;

    // workspace layout (sums f32, then 32B-aligned packed f16 weights)
    char* ws = (char*)d_ws;
    float* sumN = (float*)ws;                       // 50*64 f32
    float* sumE = sumN + N_GRAPHS * ND;             // 50*32 f32
    _Float16* pw1n = (_Float16*)(ws + 20480);       // 8nt*6kt*512 = 24576 h
    _Float16* pw2n = pw1n + 8 * 6 * 512;            // 4nt*4kt*512 =  8192 h
    _Float16* pw1e = pw2n + 4 * 4 * 512;            // 4nt*6kt*512 = 12288 h
    _Float16* pw2e = pw1e + 4 * 6 * 512;            // 2nt*2kt*512 =  2048 h

    zero_kernel<<<2048, 256, 0, stream>>>(outNodes, sumN);
    prep_weights<<<64, 256, 0, stream>>>(nW1, nW2, eW1, eW2,
                                         pw1n, pw2n, pw1e, pw2e);
    sum_nodes_kernel<<<N_GRAPHS, 256, 0, stream>>>(nodes, sumN);
    sum_edges_kernel<<<N_GRAPHS, 256, 0, stream>>>(edges, ew, sumE);
    edge_mlp_kernel<<<N_EDGES / TILE_E, 256, 0, stream>>>(
        nodes, edges, globals_, ew, senders, receivers,
        pw1n, nb1, pw2n, nb2, pw1e, eb1, pw2e, eb2,
        outNodes, outEdges);
    globals_kernel<<<N_GRAPHS, 64, 0, stream>>>(
        sumN, sumE, globals_,
        gnW1, gnb1, gnW2, gnb2, geW1, geb1, geW2, geb2,
        ggW1, ggb1, ggW2, ggb2, gfW1, gfb1, gfW2, gfb2, outG);
}
